// DGCNN_15925738733753
// MI455X (gfx1250) — compile-verified, matrix operand order checked by reading, not run on hardware
//
#include <hip/hip_runtime.h>
#include <hip/hip_bf16.h>
#include <float.h>

#define BB   32
#define NN   1024
#define KNB  20
#define EPSBN 1e-5f

typedef __attribute__((ext_vector_type(16))) _Float16 v16h;
typedef __attribute__((ext_vector_type(2)))  _Float16 v2h;
typedef __attribute__((ext_vector_type(8)))  float    v8f;
typedef __attribute__((ext_vector_type(2)))  float    v2f;

#if __has_builtin(__builtin_amdgcn_wmma_f32_16x16x4_f32)
#define HAVE_WMMA_F32 1
#else
#define HAVE_WMMA_F32 0
#endif

// K-index pattern for 16-bit WMMA A/B fragments (ISA 7.12.2, 16-bit A 16x32):
// lanes 0-15: VGPR0..3 -> K=0..7, VGPR4..7 -> K=16..23; lanes 16-31: +8.
__device__ __host__ __forceinline__ int kpat(int h, int grp) {
    int v  = h >> 1, pr = h & 1;
    int kb = (v < 4) ? (2 * v) : (16 + 2 * (v - 4));
    return kb + 8 * grp + pr;
}

// ---------------------------------------------------------------------------
// KNN: one block = one 16-row tile of pd; 16 waves.
// Phase 1: inner products via V_WMMA_F32_16X16X4_F32 (4 col-tiles per wave).
//   pd ordering trick: 2*inner - xx_n - xx_m  ==  (2*inner - xx_m) + const(row)
//   -> row norms dropped (per-row top-k invariant). Col norms folded from the
//   B fragments themselves (+ shfl_xor(16) for the partner K-half).
// Phase 2: per-wave top-20 via shfl_xor butterfly argmax, lowest-index ties
//   (matches lax.top_k); zero barriers in the selection loop.
// ---------------------------------------------------------------------------
template <int C>
__global__ void knn_kernel(const float* __restrict__ feat, int* __restrict__ idxout) {
    __shared__ float pd[16 * NN];                 // 64 KB
    const int w    = threadIdx.x >> 5;            // 0..15
    const int lane = threadIdx.x & 31;
    const int grp  = lane >> 4, l15 = lane & 15;
    const int b    = blockIdx.y;
    const int row0 = blockIdx.x * 16;
    const float* fb = feat + (size_t)b * NN * C;

#if HAVE_WMMA_F32
    constexpr int KS = (C + 3) / 4;
    // A fragments (rows row0..row0+15), held across all column tiles.
    // 32-bit A 16x4 layout: lane holds M=l15; K = 2*grp + e (ISA 7.12.2).
    v2f a[KS];
    {
        const float* ar = fb + (size_t)(row0 + l15) * C;
#pragma unroll
        for (int kk = 0; kk < KS; ++kk)
#pragma unroll
            for (int e = 0; e < 2; ++e) {
                const int k = kk * 4 + 2 * grp + e;
                a[kk][e] = (k < C) ? ar[k] : 0.0f;
            }
    }
#pragma unroll
    for (int q = 0; q < 4; ++q) {
        const int colbase = (w * 4 + q) * 16;
        const float* br = fb + (size_t)(colbase + l15) * C;
        v8f  acc = {};
        float nrm = 0.0f;
#pragma unroll
        for (int kk = 0; kk < KS; ++kk) {
            v2f bf;
#pragma unroll
            for (int e = 0; e < 2; ++e) {
                const int k = kk * 4 + 2 * grp + e;
                bf[e] = (k < C) ? br[k] : 0.0f;
            }
            nrm += bf[0] * bf[0] + bf[1] * bf[1];
            acc = __builtin_amdgcn_wmma_f32_16x16x4_f32(false, a[kk], false, bf,
                                                        (short)0, acc, false, false);
        }
        const float xxc = nrm + __shfl_xor(nrm, 16, 32);   // full column norm
#pragma unroll
        for (int r = 0; r < 8; ++r)                         // D row = r + 8*grp
            pd[(r + 8 * grp) * NN + colbase + l15] = 2.0f * acc[r] - xxc;
    }
#else
    // Scalar fallback: wave w computes -||x[row0+w]-x[m]||^2 for its own row.
    {
        const float* ctr = fb + (size_t)(row0 + w) * C;
        float cx[C];
#pragma unroll
        for (int c = 0; c < C; ++c) cx[c] = ctr[c];
        for (int j = 0; j < NN / 32; ++j) {
            const int m = lane + 32 * j;
            const float* fm = fb + (size_t)m * C;
            float s = 0.0f;
#pragma unroll
            for (int c = 0; c < C; ++c) { float d = cx[c] - fm[c]; s -= d * d; }
            pd[w * NN + m] = s;
        }
    }
#endif
    __syncthreads();

    // top-20 for local row w
    int myTop = 0;
    for (int it = 0; it < KNB; ++it) {
        float bv = -FLT_MAX; int bi = NN;
        for (int j = 0; j < NN / 32; ++j) {
            const int m = lane + 32 * j;
            const float v = pd[(size_t)w * NN + m];
            if (v > bv || (v == bv && m < bi)) { bv = v; bi = m; }
        }
#pragma unroll
        for (int off = 16; off > 0; off >>= 1) {            // butterfly: all converge
            const float ov = __shfl_xor(bv, off, 32);
            const int   oi = __shfl_xor(bi, off, 32);
            if (ov > bv || (ov == bv && oi < bi)) { bv = ov; bi = oi; }
        }
        if (lane == it) myTop = bi;
        pd[(size_t)w * NN + bi] = -FLT_MAX;  // all lanes, same addr/value: self-ordered
    }
    if (lane < KNB) idxout[((size_t)b * NN + row0 + w) * KNB + lane] = myTop;
}

// ---------------------------------------------------------------------------
// Pre-swizzle weights (CO, 2C) f32 -> f16 in WMMA B-fragment order:
// out[((ct*KT+kk)*32 + lane)*16 + h] = W[ct*16 + (lane&15)][kk*32 + kpat(h,lane>>4)]
// ---------------------------------------------------------------------------
__global__ void wswz_kernel(const float* __restrict__ W, _Float16* __restrict__ out,
                            int twoC, int KT) {
    const int lane = threadIdx.x;
    const int seg  = blockIdx.x;                  // ct*KT + kk
    const int ct = seg / KT, kk = seg % KT;
    const int grp = lane >> 4, l15 = lane & 15;
    const int o = ct * 16 + l15;
#pragma unroll
    for (int h = 0; h < 16; ++h) {
        const int k = kk * 32 + kpat(h, grp);
        out[((size_t)seg * 32 + lane) * 16 + h] =
            (k < twoC) ? (_Float16)W[(size_t)o * twoC + k] : (_Float16)0.0f;
    }
}

// ---------------------------------------------------------------------------
// EdgeConv via WMMA f16->f32. One WAVE per (b, point); 4 waves/block, no barriers.
// 32 edge rows per point (20 neighbors + duplicates of #19; max-invariant).
// Edge tile written to LDS pre-swizzled into A-fragment order (paired-half b32
// stores); A fragments are contiguous 32B ds loads, B fragments 32B global loads.
// Epilogue fuses BN + ReLU + max over all 32 rows.
// ---------------------------------------------------------------------------
#define WPB_CV 4
template <int C, int CO, int KT>
__global__ void edgeconv_wmma_kernel(const float* __restrict__ feat,
                                     const int*   __restrict__ nidx,
                                     const _Float16* __restrict__ wsw,
                                     const float* __restrict__ gam,
                                     const float* __restrict__ bet,
                                     float* __restrict__ out) {
    constexpr int twoC = 2 * C;
    constexpr int KPAD = KT * 32;
    __shared__ __align__(32) _Float16 sA[WPB_CV * 2 * 4 * 32 * 16];  // 32 KB max

    const int w    = threadIdx.x >> 5;
    const int lane = threadIdx.x & 31;
    const int p    = blockIdx.x * WPB_CV + w;
    const int b    = blockIdx.y;
    const int l15  = lane & 15;

    _Float16* sAw = sA + (size_t)w * 4096;
    const size_t rowbase = (size_t)b * NN + p;
    const float* ctr = feat + rowbase * C;

    // Build row r = lane of the 32 x KPAD edge tile, swizzled, 2 halfs at a time.
    {
        const int r   = lane;
        const int nb  = (r < KNB) ? r : (KNB - 1);
        const int src = nidx[rowbase * KNB + nb];
        const float* fs = feat + ((size_t)b * NN + src) * C;
        const int t = r >> 4, m = r & 15;
        auto edgeval = [&](int c) -> float {
            if (c < C)        return fs[c] - ctr[c];
            else if (c < twoC) return ctr[c - C];
            else               return 0.0f;
        };
#pragma unroll
        for (int c2 = 0; c2 < KPAD / 2; ++c2) {
            const int c0 = 2 * c2;
            const int kk  = c0 >> 5, kl = c0 & 31;
            const int grp2 = (kl >> 3) & 1;
            const int klr  = kl - 8 * grp2;
            const int vv   = (klr < 8) ? (klr >> 1) : (4 + ((klr - 16) >> 1));
            v2h pr2;
            pr2[0] = (_Float16)edgeval(c0);
            pr2[1] = (_Float16)edgeval(c0 + 1);
            *reinterpret_cast<v2h*>(
                &sAw[((size_t)(t * KT + kk) * 32 + (grp2 * 16 + m)) * 16 + 2 * vv]) = pr2;
        }
    }
    // Same-wave LDS store->load: in-order per wave, no barrier needed.

    const v16h* sAv = reinterpret_cast<const v16h*>(sAw);
    const v16h* wv  = reinterpret_cast<const v16h*>(wsw);
    const float invs = rsqrtf(1.0f + EPSBN);

#pragma unroll
    for (int ct = 0; ct < (CO >> 4); ++ct) {
        v8f acc0 = {}, acc1 = {};
        const int o = ct * 16 + l15;
#pragma unroll
        for (int kk = 0; kk < KT; ++kk) {
            const v16h bf = wv[(size_t)(ct * KT + kk) * 32 + lane];   // 2x b128
            const v16h a0 = sAv[(size_t)(0 * KT + kk) * 32 + lane];   // 2x ds b128
            const v16h a1 = sAv[(size_t)(1 * KT + kk) * 32 + lane];
            acc0 = __builtin_amdgcn_wmma_f32_16x16x32_f16(false, a0, false, bf,
                                                          (short)0, acc0, false, false);
            acc1 = __builtin_amdgcn_wmma_f32_16x16x32_f16(false, a1, false, bf,
                                                          (short)0, acc1, false, false);
        }
        const float scale = gam[o] * invs;
        const float bias  = bet[o];
        float mx = 0.0f;                    // ReLU floor: max(relu(h)) == max(0, max h)
#pragma unroll
        for (int r = 0; r < 8; ++r) {
            mx = fmaxf(mx, acc0[r] * scale + bias);
            mx = fmaxf(mx, acc1[r] * scale + bias);
        }
        mx = fmaxf(mx, __shfl_xor(mx, 16, 32));   // merge the two 8-row lane groups
        if (lane < 16) out[rowbase * CO + o] = mx;
    }
}

// ---------------------------------------------------------------------------
// Global max-pool over points -> (B, 320) concat [x1|x2|x3|x4].
// ---------------------------------------------------------------------------
__global__ void pool_kernel(const float* __restrict__ x1, const float* __restrict__ x2,
                            const float* __restrict__ x3, const float* __restrict__ x4,
                            float* __restrict__ pooled) {
    const int b = blockIdx.x, c = threadIdx.x;   // block = 320 threads
    const float* src; int Cc, cl;
    if      (c <  64) { src = x1; Cc = 64;  cl = c;       }
    else if (c < 128) { src = x2; Cc = 64;  cl = c - 64;  }
    else if (c < 192) { src = x3; Cc = 64;  cl = c - 128; }
    else              { src = x4; Cc = 128; cl = c - 192; }
    const float* base = src + ((size_t)b * NN) * Cc + cl;
    float mx = -FLT_MAX;
    for (int p = 0; p < NN; ++p) mx = fmaxf(mx, base[(size_t)p * Cc]);
    pooled[b * 320 + c] = mx;
}

// ---------------------------------------------------------------------------
// Dense FC (+ optional BN+ReLU). ~56 MFLOP total -> scalar is fine.
// ---------------------------------------------------------------------------
__global__ void fc_kernel(const float* __restrict__ in, const float* __restrict__ W,
                          const float* __restrict__ Lb, const float* __restrict__ gam,
                          const float* __restrict__ bet, float* __restrict__ out,
                          int I, int O, int bnrelu) {
    __shared__ float sin_[1024];
    const int b = blockIdx.y, t = threadIdx.x;
    for (int i = t; i < I; i += 256) sin_[i] = in[(size_t)b * I + i];
    __syncthreads();
    const int o = blockIdx.x * 256 + t;
    if (o < O) {
        float s = Lb[o];
        const float* wr = W + (size_t)o * I;
        for (int i = 0; i < I; ++i) s += sin_[i] * wr[i];
        if (bnrelu) s = fmaxf(s * (gam[o] * rsqrtf(1.0f + EPSBN)) + bet[o], 0.0f);
        out[(size_t)b * O + o] = s;
    }
}

// ---------------------------------------------------------------------------
extern "C" void kernel_launch(void* const* d_in, const int* in_sizes, int n_in,
                              void* d_out, int out_size, void* d_ws, size_t ws_size,
                              hipStream_t stream) {
    (void)in_sizes; (void)n_in; (void)out_size; (void)ws_size;
    const float* x   = (const float*)d_in[0];
    const float* w1  = (const float*)d_in[1];  const float* g1 = (const float*)d_in[2];  const float* b1 = (const float*)d_in[3];
    const float* w2  = (const float*)d_in[4];  const float* g2 = (const float*)d_in[5];  const float* b2 = (const float*)d_in[6];
    const float* w3  = (const float*)d_in[7];  const float* g3 = (const float*)d_in[8];  const float* b3 = (const float*)d_in[9];
    const float* w4  = (const float*)d_in[10]; const float* g4 = (const float*)d_in[11]; const float* b4 = (const float*)d_in[12];
    const float* lw1 = (const float*)d_in[13]; const float* lb1 = (const float*)d_in[14];
    const float* g5  = (const float*)d_in[15]; const float* b5  = (const float*)d_in[16];
    const float* lw2 = (const float*)d_in[17]; const float* lb2 = (const float*)d_in[18];
    const float* g6  = (const float*)d_in[19]; const float* b6  = (const float*)d_in[20];
    const float* lw3 = (const float*)d_in[21]; const float* lb3 = (const float*)d_in[22];

    char* wsp = (char*)d_ws;
    auto alloc = [&](size_t bytes) {
        char* p = wsp; wsp += (bytes + 255) & ~(size_t)255; return p;
    };
    float* x1     = (float*)alloc((size_t)BB * NN * 64  * sizeof(float));
    float* x2     = (float*)alloc((size_t)BB * NN * 64  * sizeof(float));
    float* x3     = (float*)alloc((size_t)BB * NN * 64  * sizeof(float));
    float* x4     = (float*)alloc((size_t)BB * NN * 128 * sizeof(float));
    int*   nidx   = (int*)  alloc((size_t)BB * NN * KNB * sizeof(int));
    float* pooled = (float*)alloc((size_t)BB * 320  * sizeof(float));
    float* h1     = (float*)alloc((size_t)BB * 1024 * sizeof(float));
    float* h2     = (float*)alloc((size_t)BB * 512  * sizeof(float));
    _Float16* w1s = (_Float16*)alloc((size_t)4 * 1 * 512 * sizeof(_Float16));
    _Float16* w2s = (_Float16*)alloc((size_t)4 * 4 * 512 * sizeof(_Float16));
    _Float16* w3s = (_Float16*)alloc((size_t)4 * 4 * 512 * sizeof(_Float16));
    _Float16* w4s = (_Float16*)alloc((size_t)8 * 4 * 512 * sizeof(_Float16));

    // weight pre-swizzle (tiny)
    wswz_kernel<<<dim3(4 * 1), dim3(32), 0, stream>>>(w1, w1s, 6,   1);
    wswz_kernel<<<dim3(4 * 4), dim3(32), 0, stream>>>(w2, w2s, 128, 4);
    wswz_kernel<<<dim3(4 * 4), dim3(32), 0, stream>>>(w3, w3s, 128, 4);
    wswz_kernel<<<dim3(8 * 4), dim3(32), 0, stream>>>(w4, w4s, 128, 4);

    const dim3 gKnn(NN / 16, BB), bKnn(512);
    const dim3 gCv (NN / WPB_CV, BB), bCv(32 * WPB_CV);

    knn_kernel<3><<<gKnn, bKnn, 0, stream>>>(x, nidx);
    edgeconv_wmma_kernel<3, 64, 1><<<gCv, bCv, 0, stream>>>(x,  nidx, w1s, g1, b1, x1);
    knn_kernel<64><<<gKnn, bKnn, 0, stream>>>(x1, nidx);
    edgeconv_wmma_kernel<64, 64, 4><<<gCv, bCv, 0, stream>>>(x1, nidx, w2s, g2, b2, x2);
    knn_kernel<64><<<gKnn, bKnn, 0, stream>>>(x2, nidx);
    edgeconv_wmma_kernel<64, 64, 4><<<gCv, bCv, 0, stream>>>(x2, nidx, w3s, g3, b3, x3);
    knn_kernel<64><<<gKnn, bKnn, 0, stream>>>(x3, nidx);
    edgeconv_wmma_kernel<64, 128, 4><<<gCv, bCv, 0, stream>>>(x3, nidx, w4s, g4, b4, x4);

    pool_kernel<<<dim3(BB), dim3(320), 0, stream>>>(x1, x2, x3, x4, pooled);

    fc_kernel<<<dim3(4, BB), dim3(256), 0, stream>>>(pooled, lw1, lb1, g5, b5, h1, 320,  1024, 1);
    fc_kernel<<<dim3(2, BB), dim3(256), 0, stream>>>(h1,     lw2, lb2, g6, b6, h2, 1024, 512,  1);
    fc_kernel<<<dim3(1, BB), dim3(256), 0, stream>>>(h2,     lw3, lb3, nullptr, nullptr,
                                                     (float*)d_out, 512, 40, 0);
}
